// CausalCrossAttention_80547816669835
// MI455X (gfx1250) — compile-verified, matrix operand order checked by reading, not run on hardware
//
#include <hip/hip_runtime.h>
#include <hip/hip_bf16.h>
#include <math.h>
#include <stdint.h>

typedef __attribute__((ext_vector_type(16))) _Float16 v16h;
typedef __attribute__((ext_vector_type(8)))  _Float16 v8h;
typedef __attribute__((ext_vector_type(8)))  float    v8f;
typedef __attribute__((ext_vector_type(4)))  int      v4i;

#if defined(__has_builtin)
#if __has_builtin(__builtin_amdgcn_global_load_async_to_lds_b128)
#define HAVE_ASYNC_LDS 1
#endif
#endif
#ifndef HAVE_ASYNC_LDS
#define HAVE_ASYNC_LDS 0
#endif

// global (AS1) and LDS (AS3) pointers to 128-bit payloads, per the builtin's
// prototype revealed by the round-2 diagnostic.
#define GAS4(p) ((__attribute__((address_space(1))) v4i*)(p))
#define LAS4(p) ((__attribute__((address_space(3))) v4i*)(p))

// ---------------------------------------------------------------------------
// WMMA helper: D = A(16x32 f16) * B(32x16 f16) + C(16x16 f32)
// ---------------------------------------------------------------------------
__device__ __forceinline__ v8f wmma_f16(v16h a, v16h b, v8f c) {
  return __builtin_amdgcn_wmma_f32_16x16x32_f16(
      false, a, false, b, (short)0, c, false, false);
}

// Per-lane fragment = two contiguous 8-halfword chunks (ISA 16-bit A layout:
// VGPR0..3 hold K = laneHi*8 + 0..7, VGPR4..7 hold K = 16 + laneHi*8 + 0..7).
__device__ __forceinline__ v16h make_frag(const _Float16* p) {
  v8h lo = *(const v8h*)(p);
  v8h hi = *(const v8h*)(p + 16);
  return __builtin_shufflevector(lo, hi, 0, 1, 2, 3, 4, 5, 6, 7,
                                         8, 9, 10, 11, 12, 13, 14, 15);
}

// ---------------------------------------------------------------------------
// f32 -> f16 bulk convert (8 elements / thread)
// ---------------------------------------------------------------------------
__global__ __launch_bounds__(256) void cvt_f32_f16(
    const float* __restrict__ src, _Float16* __restrict__ dst, int n) {
  int i = (blockIdx.x * 256 + threadIdx.x) * 8;
  if (i < n) {
    float4 a = *(const float4*)(src + i);
    float4 b = *(const float4*)(src + i + 4);
    v8h o;
    o[0] = (_Float16)a.x; o[1] = (_Float16)a.y; o[2] = (_Float16)a.z; o[3] = (_Float16)a.w;
    o[4] = (_Float16)b.x; o[5] = (_Float16)b.y; o[6] = (_Float16)b.z; o[7] = (_Float16)b.w;
    *(v8h*)(dst + i) = o;
  }
}

// ---------------------------------------------------------------------------
// Transpose + convert: W[K,N] f32 -> Wt[N,K] f16 (32x32 LDS tiles, coalesced)
// ---------------------------------------------------------------------------
__global__ __launch_bounds__(256) void transpose_cvt(
    const float* __restrict__ W, _Float16* __restrict__ Wt, int K, int N) {
  __shared__ float tile[32][33];
  const int k0 = blockIdx.y * 32, n0 = blockIdx.x * 32;
  const int t = threadIdx.x;
#pragma unroll
  for (int i = 0; i < 4; ++i) {
    int idx = t + i * 256;
    int r = idx >> 5, c = idx & 31;
    tile[r][c] = W[(size_t)(k0 + r) * N + n0 + c];
  }
  __syncthreads();
#pragma unroll
  for (int i = 0; i < 4; ++i) {
    int idx = t + i * 256;
    int r = idx >> 5, c = idx & 31;
    Wt[(size_t)(n0 + r) * K + k0 + c] = (_Float16)tile[c][r];
  }
}

// ---------------------------------------------------------------------------
// GEMM: C[M,N] = A[M,K] * Wt[N,K]^T + bias[N]  (f16 operands, f32 accumulate)
// 128x128x64 block tile, double-buffered async global->LDS staging.
// 8 waves (2x4); each wave 64x32 output = 4x2 WMMA tiles, 16 WMMA per stage.
// ---------------------------------------------------------------------------
#define GBM 128
#define GBN 128
#define GBK 64
#define LDT 72   // padded halfword stride: 16B-aligned, conflict-free frags

template <bool OUT_F16>
__global__ __launch_bounds__(256) void wmma_gemm_f16(
    const _Float16* __restrict__ A,   // [M,K]
    const _Float16* __restrict__ Wt,  // [N,K]  (pre-transposed weights)
    const float* __restrict__ bias,   // [N]
    void* __restrict__ Cout, int M, int N, int K) {
  __shared__ _Float16 As[2][GBM * LDT];
  __shared__ _Float16 Bs[2][GBN * LDT];

  const int t      = threadIdx.x;
  const int lane   = t & 31;
  const int wave   = t >> 5;
  const int wm     = wave >> 2;   // 0..1
  const int wn     = wave & 3;    // 0..3
  const int laneHi = lane >> 4;
  const int laneLo = lane & 15;
  const int blockM = blockIdx.y * GBM;
  const int blockN = blockIdx.x * GBN;

  auto stage = [&](int bufi, int k0) {
#pragma unroll
    for (int i = 0; i < 4; ++i) {
      int idx = t + i * 256;            // 0..1023
      int row = idx >> 3;               // 0..127
      int c8  = (idx & 7) * 8;          // halfword offset 0..56
      const _Float16* ga = A  + (size_t)(blockM + row) * K + k0 + c8;
      const _Float16* gb = Wt + (size_t)(blockN + row) * K + k0 + c8;
#if HAVE_ASYNC_LDS
      __builtin_amdgcn_global_load_async_to_lds_b128(
          GAS4(ga), LAS4(&As[bufi][row * LDT + c8]), 0, 0);
      __builtin_amdgcn_global_load_async_to_lds_b128(
          GAS4(gb), LAS4(&Bs[bufi][row * LDT + c8]), 0, 0);
#else
      *(v8h*)&As[bufi][row * LDT + c8] = *(const v8h*)ga;
      *(v8h*)&Bs[bufi][row * LDT + c8] = *(const v8h*)gb;
#endif
    }
  };

  v8f acc[4][2] = {};
  const int nk = K / GBK;
  stage(0, 0);

  for (int kt = 0; kt < nk; ++kt) {
    if (kt + 1 < nk) {
      stage((kt + 1) & 1, (kt + 1) * GBK);   // prefetch next tile (other buf)
#if HAVE_ASYNC_LDS
      asm volatile("s_wait_asynccnt 0x8" ::: "memory");  // tile kt complete
#endif
    } else {
#if HAVE_ASYNC_LDS
      asm volatile("s_wait_asynccnt 0x0" ::: "memory");
#endif
    }
    __syncthreads();

    const _Float16* as = As[kt & 1];
    const _Float16* bs = Bs[kt & 1];
#pragma unroll
    for (int kk = 0; kk < 2; ++kk) {
      v16h bfrag[2];
#pragma unroll
      for (int nt = 0; nt < 2; ++nt)
        bfrag[nt] = make_frag(&bs[(wn * 32 + nt * 16 + laneLo) * LDT + kk * 32 + laneHi * 8]);
#pragma unroll
      for (int mt = 0; mt < 4; ++mt) {
        v16h afrag = make_frag(&as[(wm * 64 + mt * 16 + laneLo) * LDT + kk * 32 + laneHi * 8]);
        acc[mt][0] = wmma_f16(afrag, bfrag[0], acc[mt][0]);
        acc[mt][1] = wmma_f16(afrag, bfrag[1], acc[mt][1]);
      }
    }
    __syncthreads();
  }

  // epilogue: C layout -> M = r + laneHi*8, N = laneLo
#pragma unroll
  for (int mt = 0; mt < 4; ++mt) {
#pragma unroll
    for (int nt = 0; nt < 2; ++nt) {
      int col = blockN + wn * 32 + nt * 16 + laneLo;
      float bv = bias[col];
#pragma unroll
      for (int r = 0; r < 8; ++r) {
        size_t row = (size_t)(blockM + wm * 64 + mt * 16 + r + laneHi * 8);
        float v = acc[mt][nt][r] + bv;
        if (OUT_F16) ((_Float16*)Cout)[row * N + col] = (_Float16)v;
        else         ((float*)Cout)[row * N + col] = v;
      }
    }
  }
}

// ---------------------------------------------------------------------------
// Flash attention (causal), f16 Q/K/V. One block per (b, h, 128-query tile).
// Wave w owns 16 query rows (private online softmax). 32-key steps:
// S = Q(16x64) K^T (2 WMMA per 16-key subtile), O += P(16x32) V (4 WMMA).
// K tile staged via async global->LDS; V staged transposed via VGPR.
// ---------------------------------------------------------------------------
#define ATT_T   2048
#define ATT_C   1024
#define ATT_H   16
#define ATT_D   64
#define LKS     72
#define LVS     40
#define LPS     40

__global__ __launch_bounds__(256) void flash_attn_kernel(
    const _Float16* __restrict__ Q,    // [B*T, C] f16
    const _Float16* __restrict__ KV,   // [B*T, 2C] f16
    _Float16* __restrict__ Y) {        // [B*T, C] f16
  __shared__ _Float16 Ks[32 * LKS];     // [s][d]
  __shared__ _Float16 Vs[64 * LVS];     // [d][s]
  __shared__ _Float16 Ps[8][16 * LPS];  // per-wave P staging

  const int t      = threadIdx.x;
  const int lane   = t & 31;
  const int wave   = t >> 5;
  const int laneHi = lane >> 4;
  const int laneLo = lane & 15;
  const int bh     = blockIdx.x;
  const int b      = bh >> 4;
  const int h      = bh & 15;
  const int qtile  = blockIdx.y;
  const int qbase  = qtile * 128 + wave * 16;
  const int qlast  = qbase + 15;
  const size_t rowb = (size_t)b * ATT_T;

  // Q fragments (f16, direct 16B loads)
  const _Float16* qp = Q + (rowb + qbase + laneLo) * ATT_C + h * ATT_D;
  v16h qfrag[2];
#pragma unroll
  for (int kk = 0; kk < 2; ++kk)
    qfrag[kk] = make_frag(qp + kk * 32 + laneHi * 8);

  float m_r[8], l_r[8];
  v8f o[4] = {};
#pragma unroll
  for (int r = 0; r < 8; ++r) { m_r[r] = -INFINITY; l_r[r] = 0.0f; }

  const _Float16* kvb = KV + rowb * (2 * ATT_C);
  const int nsteps = qtile * 4 + 4;

  for (int it = 0; it < nsteps; ++it) {
    const int s0 = it * 32;
    {
      int srow = t >> 3;              // 0..31
      int c8   = (t & 7) * 8;         // 0..56
      const _Float16* ks = kvb + (size_t)(s0 + srow) * (2 * ATT_C) + h * ATT_D + c8;
#if HAVE_ASYNC_LDS
      __builtin_amdgcn_global_load_async_to_lds_b128(
          GAS4(ks), LAS4(&Ks[srow * LKS + c8]), 0, 0);
#else
      *(v8h*)&Ks[srow * LKS + c8] = *(const v8h*)ks;
#endif
      v8h vv = *(const v8h*)(ks + ATT_C);      // V row chunk
#pragma unroll
      for (int j = 0; j < 8; ++j) Vs[(c8 + j) * LVS + srow] = vv[j];
#if HAVE_ASYNC_LDS
      asm volatile("s_wait_asynccnt 0x0" ::: "memory");
#endif
    }
    __syncthreads();

    if (s0 <= qlast) {  // wave-uniform skip of fully-masked tiles
      v8f s_acc[2];
#pragma unroll
      for (int hs = 0; hs < 2; ++hs) {
        const _Float16* kp = &Ks[(hs * 16 + laneLo) * LKS + laneHi * 8];
        v16h kf0 = make_frag(kp);
        v16h kf1 = make_frag(kp + 32);
        v8f sa = {0.f, 0.f, 0.f, 0.f, 0.f, 0.f, 0.f, 0.f};
        sa = wmma_f16(qfrag[0], kf0, sa);
        sa = wmma_f16(qfrag[1], kf1, sa);
        s_acc[hs] = sa;
      }

#pragma unroll
      for (int r = 0; r < 8; ++r) {
        int qrow = qbase + r + laneHi * 8;
        float v0 = s_acc[0][r] * 0.125f;
        float v1 = s_acc[1][r] * 0.125f;
        if (s0 + laneLo > qrow)      v0 = -INFINITY;
        if (s0 + 16 + laneLo > qrow) v1 = -INFINITY;
        float mx = fmaxf(v0, v1);
#pragma unroll
        for (int off = 1; off < 16; off <<= 1)
          mx = fmaxf(mx, __shfl_xor(mx, off));
        float nm = fmaxf(m_r[r], mx);
        float p0 = __expf(v0 - nm);
        float p1 = __expf(v1 - nm);
        float rs = p0 + p1;
#pragma unroll
        for (int off = 1; off < 16; off <<= 1)
          rs += __shfl_xor(rs, off);
        float alpha = __expf(m_r[r] - nm);
        l_r[r] = l_r[r] * alpha + rs;
        m_r[r] = nm;
#pragma unroll
        for (int nt = 0; nt < 4; ++nt) o[nt][r] *= alpha;
        int mrow = r + laneHi * 8;
        Ps[wave][mrow * LPS + laneLo]      = (_Float16)p0;
        Ps[wave][mrow * LPS + 16 + laneLo] = (_Float16)p1;
      }

      asm volatile("s_wait_dscnt 0" ::: "memory");  // same-wave LDS RAW

      const _Float16* pp = &Ps[wave][laneLo * LPS + laneHi * 8];
      v16h pfrag = make_frag(pp);
#pragma unroll
      for (int nt = 0; nt < 4; ++nt) {
        v16h vfrag = make_frag(&Vs[(nt * 16 + laneLo) * LVS + laneHi * 8]);
        o[nt] = wmma_f16(pfrag, vfrag, o[nt]);
      }
    }
    __syncthreads();
  }

  float inv_l[8];
#pragma unroll
  for (int r = 0; r < 8; ++r) inv_l[r] = 1.0f / l_r[r];
#pragma unroll
  for (int nt = 0; nt < 4; ++nt) {
#pragma unroll
    for (int r = 0; r < 8; ++r) {
      size_t row = rowb + qbase + r + laneHi * 8;
      Y[row * ATT_C + h * ATT_D + nt * 16 + laneLo] =
          (_Float16)(o[nt][r] * inv_l[r]);
    }
  }
}

// ---------------------------------------------------------------------------
// Pipeline:
//   cvt x, context -> f16 ; transpose+cvt Wq, Wkv, Wp -> f16 [N,K]
//   GEMM Q (f16 out), GEMM KV (f16 out), flash attention, GEMM out (f32 out)
// Workspace (f16): x16 8M | ctx16 8M | WqT 2M | WkvT 4M | WpT 2M |
//                  Q16 8M | KV16 16M | Y16 8M   = 56 MB
// ---------------------------------------------------------------------------
extern "C" void kernel_launch(void* const* d_in, const int* in_sizes, int n_in,
                              void* d_out, int out_size, void* d_ws, size_t ws_size,
                              hipStream_t stream) {
  const float* x       = (const float*)d_in[0];
  const float* context = (const float*)d_in[1];
  const float* Wq      = (const float*)d_in[2];
  const float* bq      = (const float*)d_in[3];
  const float* Wkv     = (const float*)d_in[4];
  const float* bkv     = (const float*)d_in[5];
  const float* Wp      = (const float*)d_in[6];
  const float* bp      = (const float*)d_in[7];
  float* out = (float*)d_out;

  const int M = 4096, C = 1024;
  char* ws = (char*)d_ws;
  const size_t MB = (size_t)1 << 20;
  _Float16* x16   = (_Float16*)(ws);
  _Float16* ctx16 = (_Float16*)(ws + 8 * MB);
  _Float16* WqT   = (_Float16*)(ws + 16 * MB);
  _Float16* WkvT  = (_Float16*)(ws + 18 * MB);
  _Float16* WpT   = (_Float16*)(ws + 22 * MB);
  _Float16* Q16   = (_Float16*)(ws + 24 * MB);
  _Float16* KV16  = (_Float16*)(ws + 32 * MB);
  _Float16* Y16   = (_Float16*)(ws + 48 * MB);

  dim3 blk(256);
  cvt_f32_f16<<<dim3((M * C) / 2048), blk, 0, stream>>>(x, x16, M * C);
  cvt_f32_f16<<<dim3((M * C) / 2048), blk, 0, stream>>>(context, ctx16, M * C);
  transpose_cvt<<<dim3(C / 32, C / 32), blk, 0, stream>>>(Wq, WqT, C, C);
  transpose_cvt<<<dim3((2 * C) / 32, C / 32), blk, 0, stream>>>(Wkv, WkvT, C, 2 * C);
  transpose_cvt<<<dim3(C / 32, C / 32), blk, 0, stream>>>(Wp, WpT, C, C);

  wmma_gemm_f16<true><<<dim3(C / GBN, M / GBM), blk, 0, stream>>>(
      x16, WqT, bq, Q16, M, C, C);
  wmma_gemm_f16<true><<<dim3((2 * C) / GBN, M / GBM), blk, 0, stream>>>(
      ctx16, WkvT, bkv, KV16, M, 2 * C, C);
  flash_attn_kernel<<<dim3(2 * ATT_H, ATT_T / 128), blk, 0, stream>>>(
      Q16, KV16, Y16);
  wmma_gemm_f16<false><<<dim3(C / GBN, M / GBM), blk, 0, stream>>>(
      Y16, WpT, bp, out, M, C, C);
}
